// MaskedSelfAttention_1726576853390
// MI455X (gfx1250) — compile-verified, matrix operand order checked by reading, not run on hardware
//
#include <hip/hip_runtime.h>
#include <hip/hip_bf16.h>

typedef unsigned short u16;
typedef unsigned int   u32;
typedef __attribute__((ext_vector_type(16))) __bf16 v16bf;
typedef __attribute__((ext_vector_type(8)))  float  v8f;

#define BATCH  4
#define TSEQ   2048
#define CDIM   768
#define NHEAD  12
#define HD     64
#define MROWS  (BATCH * TSEQ)        /* 8192 */
#define WELEM  (CDIM * CDIM)         /* 589824 */
#define MKELEM ((size_t)MROWS * CDIM)/* 6291456 */

__device__ __forceinline__ u16 f2bf(float f) {
  u32 u = __float_as_uint(f);
  u += 0x7FFFu + ((u >> 16) & 1u);   // round-to-nearest-even
  return (u16)(u >> 16);
}

// Load a 16-element bf16 fragment in the CDNA5 16-bit A/B striping:
// lanes 0-15 take K {kbase+0..7, kbase+16..23}, lanes 16-31 take {kbase+8..15, kbase+24..31}.
__device__ __forceinline__ v16bf load_frag(const u16* p, int stride, int row, int kbase, int lane) {
  int hi = (lane >> 4) & 1;
  const u16* base = p + (size_t)row * stride + kbase + hi * 8;
  union { v16bf v; uint4 q[2]; } f;
  f.q[0] = *reinterpret_cast<const uint4*>(base);
  f.q[1] = *reinterpret_cast<const uint4*>(base + 16);
  return f.v;
}

// ---------------- elementwise f32 -> bf16 ----------------
__global__ void cvt_bf16_kernel(const float* __restrict__ s, u16* __restrict__ d, int n) {
  int i = blockIdx.x * 256 + threadIdx.x;
  if (i < n) d[i] = f2bf(s[i]);
}

// W [K,N] f32  ->  Wt [N,K] bf16   (so GEMM B-fragments are contiguous)
__global__ void transpose_cvt_kernel(const float* __restrict__ W, u16* __restrict__ Wt, int K, int N) {
  int i = blockIdx.x * 256 + threadIdx.x;
  if (i >= K * N) return;
  int n = i / K;
  int k = i - n * K;
  Wt[i] = f2bf(W[(size_t)k * N + n]);
}

// v [B,T,C] bf16 -> vT [B,H,HD,T] bf16 (so P*V B-fragments are contiguous)
__global__ void transpose_v_kernel(const u16* __restrict__ v, u16* __restrict__ vT) {
  int i = blockIdx.x * 256 + threadIdx.x;
  if (i >= (int)(BATCH * NHEAD * HD * TSEQ)) return;
  int t = i % TSEQ; int r = i / TSEQ;
  int d = r % HD;   r /= HD;
  int h = r % NHEAD; int b = r / NHEAD;
  vT[i] = v[((size_t)(b * TSEQ + t)) * CDIM + h * HD + d];
}

// ---------------- bf16 WMMA GEMM:  out[M,N] = A[M,K] * Bt[N,K]^T + bias ----------------
// block = 128 threads = 4 waves (2x2) -> 128x128 block tile; each wave computes a
// 64x64 register tile (4x4 of 16x16), double-buffering the next K-step's fragments.
__global__ __launch_bounds__(128) void wmma_gemm_bias_kernel(
    const u16* __restrict__ A, const u16* __restrict__ Bt, const float* __restrict__ bias,
    float* __restrict__ outF, u16* __restrict__ outB, int M, int N, int K) {
  int lane = threadIdx.x & 31;
  int wave = threadIdx.x >> 5;
  int l15  = lane & 15;
  int hi   = (lane >> 4) & 1;
  int tileM0 = blockIdx.x * 128 + (wave >> 1) * 64;   // wave's 64-row band
  int tileN0 = blockIdx.y * 128 + (wave & 1) * 64;    // wave's 64-col band

  int arow[4], brow[4];
#pragma unroll
  for (int t = 0; t < 4; ++t) {
    arow[t] = tileM0 + t * 16 + l15;
    brow[t] = tileN0 + t * 16 + l15;
  }

  v8f acc[4][4] = {};
  v16bf a[4], b[4];

  // prologue: fragments for k = 0
#pragma unroll
  for (int t = 0; t < 4; ++t) {
    a[t] = load_frag(A,  K, arow[t], 0, lane);
    b[t] = load_frag(Bt, K, brow[t], 0, lane);
  }

  for (int k = 32; k < K; k += 32) {
    v16bf an[4], bn[4];
#pragma unroll
    for (int t = 0; t < 4; ++t) {                     // issue next K-step loads first
      an[t] = load_frag(A,  K, arow[t], k, lane);
      bn[t] = load_frag(Bt, K, brow[t], k, lane);
    }
#pragma unroll
    for (int mt = 0; mt < 4; ++mt)                    // 16 WMMAs on current fragments
#pragma unroll
      for (int nt = 0; nt < 4; ++nt)
        acc[mt][nt] = __builtin_amdgcn_wmma_f32_16x16x32_bf16(
            false, a[mt], false, b[nt], (short)0, acc[mt][nt], false, false);
#pragma unroll
    for (int t = 0; t < 4; ++t) { a[t] = an[t]; b[t] = bn[t]; }
  }
#pragma unroll
  for (int mt = 0; mt < 4; ++mt)                      // epilogue WMMAs
#pragma unroll
    for (int nt = 0; nt < 4; ++nt)
      acc[mt][nt] = __builtin_amdgcn_wmma_f32_16x16x32_bf16(
          false, a[mt], false, b[nt], (short)0, acc[mt][nt], false, false);

  // store (C/D layout: VGPR r holds M=r for lanes 0-15, M=8+r for lanes 16-31)
  float bval[4];
#pragma unroll
  for (int nt = 0; nt < 4; ++nt)
    bval[nt] = bias ? bias[tileN0 + nt * 16 + l15] : 0.0f;
#pragma unroll
  for (int mt = 0; mt < 4; ++mt) {
#pragma unroll
    for (int r = 0; r < 8; ++r) {
      int m = tileM0 + mt * 16 + r + hi * 8;
#pragma unroll
      for (int nt = 0; nt < 4; ++nt) {
        int n = tileN0 + nt * 16 + l15;
        float val = acc[mt][nt][r] + bval[nt];
        if (outF) outF[(size_t)m * N + n] = val;
        if (outB) outB[(size_t)m * N + n] = f2bf(val);
      }
    }
  }
}

// ---------------- causal flash attention (bf16 WMMA, online softmax) ----------------
// block = 128 threads = 4 waves; each wave owns one 16-row query tile; 32-key blocks.
// K fragments are double-buffered across key blocks; V fragment loads are hoisted
// above the softmax so global latency hides under the VALU-heavy reductions.
__global__ __launch_bounds__(128) void flash_attn_kernel(
    const u16* __restrict__ q, const u16* __restrict__ k, const u16* __restrict__ vT,
    u16* __restrict__ y) {
  int lane = threadIdx.x & 31;
  int wave = threadIdx.x >> 5;
  int l15  = lane & 15;
  int hi   = (lane >> 4) & 1;
  int h = blockIdx.y;
  int b = blockIdx.z;
  int q0 = (blockIdx.x * 4 + wave) * 16;

  const u16* qp = q  + (size_t)b * TSEQ * CDIM + h * HD;
  const u16* kp = k  + (size_t)b * TSEQ * CDIM + h * HD;
  const u16* vp = vT + ((size_t)b * NHEAD + h) * (size_t)(HD * TSEQ);

  __shared__ u16 lds[4][16][32];               // per-wave 16x32 bf16 P tile

  // Q fragments for the whole tile (K = 0..31 and 32..63 of head dim)
  v16bf qf0 = load_frag(qp, CDIM, q0 + l15, 0,  lane);
  v16bf qf1 = load_frag(qp, CDIM, q0 + l15, 32, lane);

  v8f acc[4] = {};                             // 16x64 output accumulator (4 N-tiles)
  float mrow[8], lrow[8];
#pragma unroll
  for (int r = 0; r < 8; ++r) { mrow[r] = -1e30f; lrow[r] = 0.0f; }

  const float scale = 0.125f;                  // 1/sqrt(64)
  const int kbEnd = q0 + 16;                   // keys <= q0+15 are the last needed

  // prologue: K fragments for kb = 0   [kf: {sub-tile 0/1} x {khalf 0/1}]
  v16bf kf[4];
  kf[0] = load_frag(kp, CDIM, 0  + l15, 0,  lane);
  kf[1] = load_frag(kp, CDIM, 0  + l15, 32, lane);
  kf[2] = load_frag(kp, CDIM, 16 + l15, 0,  lane);
  kf[3] = load_frag(kp, CDIM, 16 + l15, 32, lane);

  for (int kb = 0; kb < kbEnd; kb += 32) {
    // ---- S = Q * K^T for key columns kb..kb+31, split into two 16x16 tiles ----
    v8f s0 = {}, s1 = {};
    s0 = __builtin_amdgcn_wmma_f32_16x16x32_bf16(false, qf0, false, kf[0], (short)0, s0, false, false);
    s0 = __builtin_amdgcn_wmma_f32_16x16x32_bf16(false, qf1, false, kf[1], (short)0, s0, false, false);
    s1 = __builtin_amdgcn_wmma_f32_16x16x32_bf16(false, qf0, false, kf[2], (short)0, s1, false, false);
    s1 = __builtin_amdgcn_wmma_f32_16x16x32_bf16(false, qf1, false, kf[3], (short)0, s1, false, false);

    // ---- issue independent loads early: V frags for this block, K frags for next ----
    v16bf vf[4];
#pragma unroll
    for (int t = 0; t < 4; ++t)
      vf[t] = load_frag(vp, TSEQ, t * 16 + l15, kb, lane);
    {
      int kb2 = (kb + 32 < kbEnd) ? (kb + 32) : kb;   // harmless re-read on last iter
      kf[0] = load_frag(kp, CDIM, kb2      + l15, 0,  lane);
      kf[1] = load_frag(kp, CDIM, kb2      + l15, 32, lane);
      kf[2] = load_frag(kp, CDIM, kb2 + 16 + l15, 0,  lane);
      kf[3] = load_frag(kp, CDIM, kb2 + 16 + l15, 32, lane);
    }

    // ---- online softmax (row reductions across the 16-lane half-wave) ----
#pragma unroll
    for (int r = 0; r < 8; ++r) {
      int qrow = q0 + r + hi * 8;
      int kc0  = kb + l15;
      int kc1  = kb + 16 + l15;
      float x0 = (kc0 <= qrow) ? s0[r] * scale : -1e30f;
      float x1 = (kc1 <= qrow) ? s1[r] * scale : -1e30f;
      float mx = fmaxf(x0, x1);
#pragma unroll
      for (int off = 1; off < 16; off <<= 1) mx = fmaxf(mx, __shfl_xor(mx, off, 32));
      float mnew = fmaxf(mrow[r], mx);
      float p0 = __expf(x0 - mnew);
      float p1 = __expf(x1 - mnew);
      float rs = p0 + p1;
#pragma unroll
      for (int off = 1; off < 16; off <<= 1) rs += __shfl_xor(rs, off, 32);
      float alpha = __expf(mrow[r] - mnew);
      mrow[r] = mnew;
      lrow[r] = lrow[r] * alpha + rs;
#pragma unroll
      for (int t = 0; t < 4; ++t) acc[t][r] *= alpha;
      // stage P (bf16) in LDS: C-layout -> memory tile
      lds[wave][r + hi * 8][l15]      = f2bf(p0);
      lds[wave][r + hi * 8][16 + l15] = f2bf(p1);
    }
    __builtin_amdgcn_wave_barrier();           // in-wave LDS write->read ordering

    // ---- reload P as an A-fragment (16x32) ----
    union { v16bf v; uint4 qv[2]; } pf;
    pf.qv[0] = *reinterpret_cast<const uint4*>(&lds[wave][l15][hi * 8]);
    pf.qv[1] = *reinterpret_cast<const uint4*>(&lds[wave][l15][hi * 8 + 16]);
    __builtin_amdgcn_wave_barrier();

    // ---- O += P * V  (V^T layout [hd,T] -> contiguous B-fragments) ----
#pragma unroll
    for (int t = 0; t < 4; ++t)
      acc[t] = __builtin_amdgcn_wmma_f32_16x16x32_bf16(false, pf.v, false, vf[t], (short)0, acc[t], false, false);
  }

  // ---- finalize: O /= l, store bf16 y [B,T,C] ----
#pragma unroll
  for (int r = 0; r < 8; ++r) {
    int qrow = q0 + r + hi * 8;
    float inv = 1.0f / lrow[r];
#pragma unroll
    for (int t = 0; t < 4; ++t) {
      y[((size_t)(b * TSEQ + qrow)) * CDIM + h * HD + t * 16 + l15] = f2bf(acc[t][r] * inv);
    }
  }
}

// ---------------- host-side orchestration ----------------
extern "C" void kernel_launch(void* const* d_in, const int* in_sizes, int n_in,
                              void* d_out, int out_size, void* d_ws, size_t ws_size,
                              hipStream_t stream) {
  const float* x  = (const float*)d_in[0];
  const float* Wq = (const float*)d_in[1];
  const float* bq = (const float*)d_in[2];
  const float* Wk = (const float*)d_in[3];
  const float* bk = (const float*)d_in[4];
  const float* Wv = (const float*)d_in[5];
  const float* bv = (const float*)d_in[6];
  const float* Wo = (const float*)d_in[7];
  const float* bo = (const float*)d_in[8];
  float* out = (float*)d_out;

  // workspace layout (bf16 halves)
  u16* ws  = (u16*)d_ws;
  u16* xb  = ws;                 // [M,K]
  u16* qb  = xb  + MKELEM;       // [M,C]
  u16* kb  = qb  + MKELEM;
  u16* vb  = kb  + MKELEM;
  u16* vT  = vb  + MKELEM;       // [B,H,HD,T]
  u16* yb  = vT  + MKELEM;       // [M,C]
  u16* Wqt = yb  + MKELEM;       // [N,K] each
  u16* Wkt = Wqt + WELEM;
  u16* Wvt = Wkt + WELEM;
  u16* Wot = Wvt + WELEM;

  // 1) convert x and weights to bf16 (weights transposed to [N,K])
  {
    int n = (int)MKELEM;
    cvt_bf16_kernel<<<dim3((n + 255) / 256), dim3(256), 0, stream>>>(x, xb, n);
    int wn = WELEM;
    dim3 wg((wn + 255) / 256);
    transpose_cvt_kernel<<<wg, dim3(256), 0, stream>>>(Wq, Wqt, CDIM, CDIM);
    transpose_cvt_kernel<<<wg, dim3(256), 0, stream>>>(Wk, Wkt, CDIM, CDIM);
    transpose_cvt_kernel<<<wg, dim3(256), 0, stream>>>(Wv, Wvt, CDIM, CDIM);
    transpose_cvt_kernel<<<wg, dim3(256), 0, stream>>>(Wo, Wot, CDIM, CDIM);
  }

  // 2) QKV projections (bf16 out, fp32 accumulate)  — 128x128 block tiles
  {
    dim3 grid(MROWS / 128, CDIM / 128);
    wmma_gemm_bias_kernel<<<grid, dim3(128), 0, stream>>>(xb, Wqt, bq, nullptr, qb, MROWS, CDIM, CDIM);
    wmma_gemm_bias_kernel<<<grid, dim3(128), 0, stream>>>(xb, Wkt, bk, nullptr, kb, MROWS, CDIM, CDIM);
    wmma_gemm_bias_kernel<<<grid, dim3(128), 0, stream>>>(xb, Wvt, bv, nullptr, vb, MROWS, CDIM, CDIM);
  }

  // 3) transpose V per head for contiguous P*V fragments
  {
    int n = BATCH * NHEAD * HD * TSEQ;
    transpose_v_kernel<<<dim3((n + 255) / 256), dim3(256), 0, stream>>>(vb, vT);
  }

  // 4) causal flash attention
  {
    dim3 grid(TSEQ / 64, NHEAD, BATCH);   // 4 query tiles (waves) per block
    flash_attn_kernel<<<grid, dim3(128), 0, stream>>>(qb, kb, vT, yb);
  }

  // 5) output projection -> fp32 d_out
  {
    dim3 grid(MROWS / 128, CDIM / 128);
    wmma_gemm_bias_kernel<<<grid, dim3(128), 0, stream>>>(yb, Wot, bo, out, nullptr, MROWS, CDIM, CDIM);
  }
}
